// GatedMultiHeadAttention_67946382623207
// MI455X (gfx1250) — matrix operand order repaired and checked by reading, not compiled, not run on hardware
//
#include <hip/hip_runtime.h>
#include <hip/hip_bf16.h>
#include <math.h>

// ---------------------------------------------------------------------------
// Gated MHA for MI455X (gfx1250): bf16 WMMA, f32 accumulate, LDS double-
// buffered GEMM + attention with async global->LDS copies (ASYNCcnt path).
// ---------------------------------------------------------------------------

typedef __bf16 bf16_t;
typedef __attribute__((ext_vector_type(16))) __bf16 bf16x16;
typedef __attribute__((ext_vector_type(8)))  __bf16 bf16x8;
typedef __attribute__((ext_vector_type(8)))  float  f32x8;
typedef __attribute__((ext_vector_type(4)))  int    v4i;

#define WMMA_BF16(a, b, c) \
  __builtin_amdgcn_wmma_f32_16x16x32_bf16(false, (a), false, (b), (short)0, (c), false, false)

#define AS1 __attribute__((address_space(1)))
#define AS3 __attribute__((address_space(3)))

#if defined(__has_builtin)
#if __has_builtin(__builtin_amdgcn_global_load_async_to_lds_b128)
#define HAVE_ASYNC_LDS 1
#endif
#if __has_builtin(__builtin_amdgcn_s_wait_asynccnt)
#define HAVE_WAIT_ASYNC 1
#endif
#endif

static constexpr int Bb    = 2;
static constexpr int Nn    = 2048;
static constexpr int Dd    = 1024;
static constexpr int Hh    = 16;
static constexpr int HDim  = 64;
static constexpr int Mrows = Bb * Nn;     // 4096
static constexpr int NQ    = Dd + Hh;     // 1040 (Q || gate)

// 16-byte global -> LDS copy (async ASYNCcnt path when available).
__device__ __forceinline__ void cp16(bf16_t* l, const bf16_t* g) {
#ifdef HAVE_ASYNC_LDS
  __builtin_amdgcn_global_load_async_to_lds_b128((AS1 v4i*)(g), (AS3 v4i*)(l),
                                                 0, 0);
#else
  *(bf16x8*)l = *(const bf16x8*)g;
#endif
}

template <int N>
__device__ __forceinline__ void wait_async() {
#ifdef HAVE_ASYNC_LDS
#ifdef HAVE_WAIT_ASYNC
  __builtin_amdgcn_s_wait_asynccnt((unsigned short)N);
#else
  asm volatile("s_wait_asynccnt %0" ::"i"(N));
#endif
#endif
}

__device__ __forceinline__ float grpmax16(float v) {
#pragma unroll
  for (int i = 1; i < 16; i <<= 1) v = fmaxf(v, __shfl_xor(v, i, 16));
  return v;
}
__device__ __forceinline__ float grpsum16(float v) {
#pragma unroll
  for (int i = 1; i < 16; i <<= 1) v += __shfl_xor(v, i, 16);
  return v;
}
__device__ __forceinline__ bf16x16 pack16(bf16x8 lo, bf16x8 hi) {
  bf16x16 r;
#pragma unroll
  for (int i = 0; i < 8; i++) { r[i] = lo[i]; r[i + 8] = hi[i]; }
  return r;
}

// ------------------------------ casts --------------------------------------
__global__ __launch_bounds__(256) void cast_bf16_kernel(
    const float* __restrict__ in, bf16_t* __restrict__ out, int n) {
  int i = blockIdx.x * 256 + threadIdx.x;
  if (i < n) out[i] = (bf16_t)in[i];
}

// W [K][Ncols] f32 (row-major)  ->  Wt [Ncols][K] bf16 (row-major)
__global__ __launch_bounds__(256) void cast_transpose_kernel(
    const float* __restrict__ W, bf16_t* __restrict__ Wt, int Kdim, int Ncols) {
  int i = blockIdx.x * 256 + threadIdx.x;
  if (i >= Kdim * Ncols) return;
  int k = i / Ncols, n = i % Ncols;
  Wt[(size_t)n * Kdim + k] = (bf16_t)W[i];
}

// ------------------------------ GEMM ---------------------------------------
// C[M][ldC](f32) = A[M][K](bf16) * Bt[N][K](bf16)^T + bias[N]
// Block: 256 thr (8 waves), block tile 128x128, K-stage 64, LDS double buffer
// filled with async global->LDS copies. Wave tile 32x64 (2x4 WMMA accums).
static constexpr int BM = 128, BN = 128, BK = 64;

__global__ __launch_bounds__(256) void gemm_bf16_kernel(
    const bf16_t* __restrict__ A, const bf16_t* __restrict__ Bt,
    const float* __restrict__ bias, float* __restrict__ C,
    int Ncols, int Kdim, int ldC) {
  __shared__ bf16_t ldsA[2][BM][BK];   // 2 x 16 KB
  __shared__ bf16_t ldsB[2][BN][BK];   // 2 x 16 KB

  const int tid  = threadIdx.x;
  const int lane = tid & 31;
  const int w    = tid >> 5;
  const int wm   = w >> 1;                 // 0..3 : 32-row slice
  const int wn   = w & 1;                  // 0..1 : 64-col slice
  const int rBlock = blockIdx.y * BM;
  const int cBlock = blockIdx.x * BN;

  // staging coords: each thread copies 32 contiguous k-elems of one A row
  // and one B row (4 x 16B async ops each).
  const int crow = tid >> 1;               // 0..127
  const int ck   = (tid & 1) * 32;         // 0 / 32
  const bf16_t* gA = A + (size_t)(rBlock + crow) * Kdim + ck;
  int gc = cBlock + crow; if (gc > Ncols - 1) gc = Ncols - 1;
  const bf16_t* gB = Bt + (size_t)gc * Kdim + ck;

  auto stage = [&](int buf, int k0) {
    const bf16_t* ga = gA + k0;
    bf16_t* la = &ldsA[buf][crow][ck];
    cp16(la + 0, ga + 0);   cp16(la + 8, ga + 8);
    cp16(la + 16, ga + 16); cp16(la + 24, ga + 24);
    const bf16_t* gb = gB + k0;
    bf16_t* lb = &ldsB[buf][crow][ck];
    cp16(lb + 0, gb + 0);   cp16(lb + 8, gb + 8);
    cp16(lb + 16, gb + 16); cp16(lb + 24, gb + 24);
  };

  const int arow  = lane & 15;
  const int ahalf = (lane >> 4) * 8;
  const int bhalf = (lane >> 4) * 16;

  f32x8 acc[2][4];
#pragma unroll
  for (int i = 0; i < 2; i++)
#pragma unroll
    for (int t = 0; t < 4; t++) acc[i][t] = (f32x8){};

  stage(0, 0);
  const int nstage = Kdim / BK;
  for (int s = 0; s < nstage; s++) {
    const int buf = s & 1;
    if (s + 1 < nstage) {          // overlap next copy with this stage's math
      stage(buf ^ 1, (s + 1) * BK);
      wait_async<8>();             // async loads complete in order: oldest 8 done
    } else {
      wait_async<0>();
    }
    __syncthreads();
#pragma unroll
    for (int kk = 0; kk < BK; kk += 32) {
      // issue ALL fragment loads first so ds latency overlaps across them
      bf16x16 af[2], bfr[4];
#pragma unroll
      for (int i = 0; i < 2; i++) {
        const bf16_t* ap = &ldsA[buf][wm * 32 + i * 16 + arow][kk];
        af[i] = *(const bf16x16*)(ap + bhalf);
      }
#pragma unroll
      for (int t = 0; t < 4; t++)
        bfr[t] = *(const bf16x16*)&ldsB[buf][wn * 64 + t * 16 + arow][kk + bhalf];
#pragma unroll
      for (int t = 0; t < 4; t++)
#pragma unroll
        for (int i = 0; i < 2; i++) acc[i][t] = WMMA_BF16(af[i], bfr[t], acc[i][t]);
    }
    __syncthreads();
  }

  const int rh = (lane >> 4) * 8;
#pragma unroll
  for (int t = 0; t < 4; t++) {
    int col = cBlock + wn * 64 + t * 16 + (lane & 15);
    if (col < Ncols) {
      float bi = bias[col];
#pragma unroll
      for (int i = 0; i < 2; i++)
#pragma unroll
        for (int j = 0; j < 8; j++) {
          int r = rBlock + wm * 32 + i * 16 + rh + j;
          C[(size_t)r * ldC + col] = acc[i][t][j] + bi;
        }
    }
  }
}

// --------------------- RMSNorm(Q,K) + repack, V transpose ------------------
__global__ __launch_bounds__(256) void qknorm_pack_kernel(
    const float* __restrict__ qg, const float* __restrict__ kraw,
    const float* __restrict__ vraw, const float* __restrict__ qn_w,
    const float* __restrict__ kn_w,
    bf16_t* __restrict__ Qn, bf16_t* __restrict__ Kn, bf16_t* __restrict__ Vt) {
  const int lane = threadIdx.x & 31;
  const int rid  = blockIdx.x * 8 + (threadIdx.x >> 5);  // (b*N + n)*H + h
  const int h    = rid & (Hh - 1);
  const int bn   = rid >> 4;
  const int n    = bn & (Nn - 1);
  const int b    = bn >> 11;
  const int e0   = lane * 2, e1 = lane * 2 + 1;

  const float* qrow = qg + (size_t)bn * NQ + h * HDim;
  float q0 = qrow[e0], q1 = qrow[e1];
  float sq = q0 * q0 + q1 * q1;
#pragma unroll
  for (int i = 1; i < 32; i <<= 1) sq += __shfl_xor(sq, i, 32);
  float qr = rsqrtf(sq * (1.0f / HDim) + 1e-6f);

  const float* krow = kraw + (size_t)bn * Dd + h * HDim;
  float k0 = krow[e0], k1 = krow[e1];
  float sk = k0 * k0 + k1 * k1;
#pragma unroll
  for (int i = 1; i < 32; i <<= 1) sk += __shfl_xor(sk, i, 32);
  float kr = rsqrtf(sk * (1.0f / HDim) + 1e-6f);

  size_t obase = ((size_t)(b * Hh + h) * Nn + n) * HDim;
  Qn[obase + e0] = (bf16_t)(q0 * qr * qn_w[e0]);
  Qn[obase + e1] = (bf16_t)(q1 * qr * qn_w[e1]);
  Kn[obase + e0] = (bf16_t)(k0 * kr * kn_w[e0]);
  Kn[obase + e1] = (bf16_t)(k1 * kr * kn_w[e1]);

  const float* vrow = vraw + (size_t)bn * Dd + h * HDim;
  size_t vb = (size_t)(b * Hh + h) * HDim;
  Vt[(vb + e0) * Nn + n] = (bf16_t)vrow[e0];
  Vt[(vb + e1) * Nn + n] = (bf16_t)vrow[e1];
}

// ------------------------------ attention ----------------------------------
// grid (N/64, B*H); block 128 = 4 waves; each wave owns 16 query rows.
// K/V chunks double-buffered in LDS: next chunk's async copies are issued
// before waiting on the current one, overlapping copy with softmax/WMMA.
__global__ __launch_bounds__(128) void attn_kernel(
    const bf16_t* __restrict__ Qn, const bf16_t* __restrict__ Kn,
    const bf16_t* __restrict__ Vt, const float* __restrict__ qg,
    bf16_t* __restrict__ Og) {
  const int bh   = blockIdx.y;
  const int b    = bh >> 4, h = bh & 15;
  const int w    = threadIdx.x >> 5;
  const int lane = threadIdx.x & 31;
  const int tid  = threadIdx.x;
  const int q0   = blockIdx.x * 64 + w * 16;

  const bf16_t* Qb = Qn + (size_t)bh * Nn * HDim;
  const bf16_t* Kb = Kn + (size_t)bh * Nn * HDim;
  const bf16_t* Vb = Vt + (size_t)bh * HDim * Nn;

  const int arow  = lane & 15;
  const int ahalf = (lane >> 4) * 8;
  const int bhalf = (lane >> 4) * 16;

  // Q A-fragments (16 rows x 64, two k-steps of 32), loaded once.
  const bf16_t* qp = Qb + (size_t)(q0 + arow) * HDim;
  bf16x16 qf[2];
#pragma unroll
  for (int kk = 0; kk < 2; kk++)
    qf[kk] = *(const bf16x16*)(qp + kk * 32 + bhalf);

  f32x8 acc[4]; float m[8], l[8];
#pragma unroll
  for (int t = 0; t < 4; t++) acc[t] = (f32x8){};
#pragma unroll
  for (int j = 0; j < 8; j++) { m[j] = -__builtin_inff(); l[j] = 0.0f; }

  __shared__ bf16_t ldsK[2][32][64];   // K rows of kv chunk   (2 x 4 KB)
  __shared__ bf16_t ldsV[2][64][32];   // Vt rows (d) x kv     (2 x 4 KB)
  __shared__ bf16_t Pl[4][16][32];     // per-wave P staging   (4 KB)

  // per-thread staging coords (128 threads, 4 async 16B ops per stage)
  const int krow = tid >> 2, kkoff = (tid & 3) * 16;   // 32 x 64
  const int vrow = tid >> 1, vkoff = (tid & 1) * 16;   // 64 x 32

  auto stageKV = [&](int buf, int kv) {
    const bf16_t* gk = Kb + (size_t)(kv + krow) * HDim + kkoff;
    cp16(&ldsK[buf][krow][kkoff], gk);
    cp16(&ldsK[buf][krow][kkoff + 8], gk + 8);
    const bf16_t* gv = Vb + (size_t)vrow * Nn + kv + vkoff;
    cp16(&ldsV[buf][vrow][vkoff], gv);
    cp16(&ldsV[buf][vrow][vkoff + 8], gv + 8);
  };

  stageKV(0, 0);
  const int nchunk = Nn / 32;
  for (int c = 0; c < nchunk; c++) {
    const int buf = c & 1;
    const int kv  = c * 32;
    if (c + 1 < nchunk) {          // overlap next K/V copy with this chunk
      stageKV(buf ^ 1, kv + 32);
      wait_async<4>();             // in-order: oldest 4 (current chunk) done
    } else {
      wait_async<0>();
    }
    __syncthreads();

    // S = Q @ K^T for 16 rows x 32 kv cols
    f32x8 s[2]; s[0] = (f32x8){}; s[1] = (f32x8){};
#pragma unroll
    for (int c2 = 0; c2 < 2; c2++)
#pragma unroll
      for (int kk = 0; kk < 2; kk++) {
        bf16x16 kf =
            *(const bf16x16*)&ldsK[buf][c2 * 16 + (lane & 15)][kk * 32 + bhalf];
        s[c2] = WMMA_BF16(qf[kk], kf, s[c2]);
      }

    // online softmax update (row = (lane>>4)*8 + j, col = lane&15)
    float p0[8], p1[8], alpha[8];
#pragma unroll
    for (int j = 0; j < 8; j++) {
      float s0 = s[0][j] * 0.125f;      // 1/sqrt(64)
      float s1 = s[1][j] * 0.125f;
      float cm = grpmax16(fmaxf(s0, s1));
      float mnew = fmaxf(m[j], cm);
      float a  = __expf(m[j] - mnew);
      float e0 = __expf(s0 - mnew);
      float e1 = __expf(s1 - mnew);
      l[j] = l[j] * a + grpsum16(e0 + e1);
      m[j] = mnew;
      alpha[j] = a; p0[j] = e0; p1[j] = e1;
    }
#pragma unroll
    for (int t = 0; t < 4; t++)
#pragma unroll
      for (int j = 0; j < 8; j++) acc[t][j] *= alpha[j];

    // P (C-layout f32) -> LDS bf16 -> A-layout fragment
    const int prh = (lane >> 4) * 8, pc = lane & 15;
#pragma unroll
    for (int j = 0; j < 8; j++) {
      Pl[w][prh + j][pc]      = (bf16_t)p0[j];
      Pl[w][prh + j][pc + 16] = (bf16_t)p1[j];
    }
    __syncthreads();

    bf16x16 pf = *(const bf16x16*)&Pl[w][arow][bhalf];
    // O += P @ V (K-dim = 32 kv chunk), 4 d-tiles of 16
#pragma unroll
    for (int t = 0; t < 4; t++) {
      bf16x16 vf = *(const bf16x16*)&ldsV[buf][t * 16 + (lane & 15)][bhalf];
      acc[t] = WMMA_BF16(pf, vf, acc[t]);
    }
    __syncthreads();
  }

  // epilogue: 1/l, sigmoid(gate), store bf16 [B*N, D]
  const int rh = (lane >> 4) * 8, col = lane & 15;
#pragma unroll
  for (int j = 0; j < 8; j++) {
    int q = q0 + rh + j;
    float invl = 1.0f / l[j];
    float g  = qg[(size_t)(b * Nn + q) * NQ + Dd + h];
    float gv = 1.0f / (1.0f + __expf(-g));
#pragma unroll
    for (int t = 0; t < 4; t++) {
      float o = acc[t][j] * invl * gv;
      Og[(size_t)(b * Nn + q) * Dd + h * HDim + t * 16 + col] = (bf16_t)o;
    }
  }
}

// ------------------------------ launcher -----------------------------------
extern "C" void kernel_launch(void* const* d_in, const int* in_sizes, int n_in,
                              void* d_out, int out_size, void* d_ws, size_t ws_size,
                              hipStream_t stream) {
  const float* x    = (const float*)d_in[0];
  const float* Wq   = (const float*)d_in[1];
  const float* bq   = (const float*)d_in[2];
  const float* Wk   = (const float*)d_in[3];
  const float* bk   = (const float*)d_in[4];
  const float* Wv   = (const float*)d_in[5];
  const float* bv   = (const float*)d_in[6];
  const float* Wo   = (const float*)d_in[7];
  const float* bo   = (const float*)d_in[8];
  const float* qn_w = (const float*)d_in[9];
  const float* kn_w = (const float*)d_in[10];
  float* out = (float*)d_out;

  char* ws = (char*)d_ws;
  size_t off = 0;
  auto take = [&](size_t bytes) -> char* {
    char* p = ws + off;
    off += (bytes + 255) & ~(size_t)255;
    return p;
  };
  bf16_t* xb    = (bf16_t*)take((size_t)Mrows * Dd * 2);
  bf16_t* WqT   = (bf16_t*)take((size_t)NQ * Dd * 2);
  bf16_t* WkT   = (bf16_t*)take((size_t)Dd * Dd * 2);
  bf16_t* WvT   = (bf16_t*)take((size_t)Dd * Dd * 2);
  bf16_t* WoT   = (bf16_t*)take((size_t)Dd * Dd * 2);
  float*  qgf   = (float*)take((size_t)Mrows * NQ * 4);
  float*  krawf = (float*)take((size_t)Mrows * Dd * 4);
  float*  vrawf = (float*)take((size_t)Mrows * Dd * 4);
  bf16_t* Qn    = (bf16_t*)take((size_t)Mrows * Dd * 2);
  bf16_t* Kn    = (bf16_t*)take((size_t)Mrows * Dd * 2);
  bf16_t* Vt    = (bf16_t*)take((size_t)Mrows * Dd * 2);
  bf16_t* Og    = (bf16_t*)take((size_t)Mrows * Dd * 2);
  (void)ws_size; (void)in_sizes; (void)n_in; (void)out_size;

  // 1) casts
  {
    int n = Mrows * Dd;
    cast_bf16_kernel<<<(n + 255) / 256, 256, 0, stream>>>(x, xb, n);
  }
  cast_transpose_kernel<<<(Dd * NQ + 255) / 256, 256, 0, stream>>>(Wq, WqT, Dd, NQ);
  cast_transpose_kernel<<<(Dd * Dd + 255) / 256, 256, 0, stream>>>(Wk, WkT, Dd, Dd);
  cast_transpose_kernel<<<(Dd * Dd + 255) / 256, 256, 0, stream>>>(Wv, WvT, Dd, Dd);
  cast_transpose_kernel<<<(Dd * Dd + 255) / 256, 256, 0, stream>>>(Wo, WoT, Dd, Dd);

  // 2) projections (WMMA, LDS double-buffered async staging)
  dim3 blk(256);
  dim3 gq((NQ + BN - 1) / BN, Mrows / BM);
  dim3 gd(Dd / BN, Mrows / BM);
  gemm_bf16_kernel<<<gq, blk, 0, stream>>>(xb, WqT, bq, qgf,   NQ, Dd, NQ);
  gemm_bf16_kernel<<<gd, blk, 0, stream>>>(xb, WkT, bk, krawf, Dd, Dd, Dd);
  gemm_bf16_kernel<<<gd, blk, 0, stream>>>(xb, WvT, bv, vrawf, Dd, Dd, Dd);

  // 3) QK rmsnorm + repack, V transpose
  qknorm_pack_kernel<<<(Mrows * Hh) / 8, 256, 0, stream>>>(
      qgf, krawf, vrawf, qn_w, kn_w, Qn, Kn, Vt);

  // 4) flash attention + gate (WMMA, double-buffered async K/V staging)
  attn_kernel<<<dim3(Nn / 64, Bb * Hh), 128, 0, stream>>>(Qn, Kn, Vt, qgf, Og);

  // 5) output projection (WMMA)
  gemm_bf16_kernel<<<gd, blk, 0, stream>>>(Og, WoT, bo, out, Dd, Dd, Dd);
}
